// Inv_Attn_89515708383992
// MI455X (gfx1250) — compile-verified, matrix operand order checked by reading, not run on hardware
//
#include <hip/hip_runtime.h>
#include <hip/hip_bf16.h>

#define BDIM  16
#define CDIM  512
#define C2DIM 256
#define NDIM  2304   // 48*48
#define PITCH 40     // tile row pitch in halfs: 80B = 5*16B -> b128-aligned rows
#define TILEH (64 * PITCH)   // 2560 halfs per 64x32 tile

typedef __attribute__((ext_vector_type(16))) __bf16 v16bf;
typedef __attribute__((ext_vector_type(8)))  float  v8f;

__device__ __forceinline__ unsigned short f2bf_bits(float f) {
    unsigned int u = __float_as_uint(f);
    u += 0x7FFFu + ((u >> 16) & 1u);   // round-to-nearest-even
    return (unsigned short)(u >> 16);
}
__device__ __forceinline__ __bf16 bits2bf(unsigned short h) {
    return __builtin_bit_cast(__bf16, h);
}

// K offset inside a 32-wide step for a bf16 WMMA A/B fragment register-pair v.
// (ISA 16-bit A 16x32 layout: VGPR0..3 -> K0..7 (+8 hi lane half), VGPR4..7 -> K16..23 (+8))
__device__ __forceinline__ int frag_kbase(int v, int g) {
    return (v < 4 ? 2 * v : 16 + 2 * (v - 4)) + 8 * g;
}

__device__ __forceinline__ v16bf load_frag(const unsigned short* T, int rowbase, int g) {
    v16bf f;
    #pragma unroll
    for (int v = 0; v < 8; ++v) {
        int kb = frag_kbase(v, g);
        f[2 * v]     = bits2bf(T[rowbase + kb]);
        f[2 * v + 1] = bits2bf(T[rowbase + kb + 1]);
    }
    return f;
}

// LDS byte offset of a generic pointer to __shared__ (aperture maps addr[31:0]).
__device__ __forceinline__ unsigned lds_off(const void* p) {
    return (unsigned)(unsigned long long)p;
}
// Async 16B copy global->LDS (ASYNCcnt-tracked, no VGPR staging).
__device__ __forceinline__ void async_b128(unsigned lds, const void* g) {
    asm volatile("global_load_async_to_lds_b128 %0, %1, off"
                 :: "v"(lds), "v"(g) : "memory");
}

// ---------------------------------------------------------------------------
// Kernel 1: fused projections.  p=0: fwT = (Wf@w+bf)^T  (N x C2, bf16)
//                               p=1: gxT = (Wg@x+bg)^T  (N x C2, bf16)
//                               p=2: hx  =  Wh@x+bh     (C x N,  bf16)
// Transposed f/g outputs make the logits GEMM operands K-contiguous so they
// can be async-copied into LDS.
// ---------------------------------------------------------------------------
__global__ __launch_bounds__(128)
void proj_kernel(const float* __restrict__ x, const float* __restrict__ w,
                 const float* __restrict__ Wf, const float* __restrict__ bf,
                 const float* __restrict__ Wg, const float* __restrict__ bg,
                 const float* __restrict__ Wh, const float* __restrict__ bh,
                 unsigned short* __restrict__ fwT, unsigned short* __restrict__ gxT,
                 unsigned short* __restrict__ hx)
{
    const int n0 = blockIdx.x * 64;
    const int o0 = blockIdx.y * 64;
    const int bz = blockIdx.z;
    const int b  = bz / 3;
    const int p  = bz % 3;

    const float* Wp; const float* bias; const float* src;
    unsigned short* dst; int O;
    if (p == 0)      { Wp = Wf; bias = bf; src = w + (size_t)b * CDIM * NDIM;
                       dst = fwT + (size_t)b * C2DIM * NDIM; O = C2DIM; }
    else if (p == 1) { Wp = Wg; bias = bg; src = x + (size_t)b * CDIM * NDIM;
                       dst = gxT + (size_t)b * C2DIM * NDIM; O = C2DIM; }
    else             { Wp = Wh; bias = bh; src = x + (size_t)b * CDIM * NDIM;
                       dst = hx + (size_t)b * CDIM * NDIM;  O = CDIM;  }
    if (o0 >= O) return;   // uniform per block

    __shared__ __align__(16) unsigned short smem[2 * TILEH];   // As | Bs, reused as epilogue stage
    unsigned short* As = smem;
    unsigned short* Bs = smem + TILEH;

    const int tid  = threadIdx.x;
    const int lane = tid & 31;
    const int wv   = tid >> 5;
    const int g    = lane >> 4;
    const int mr   = lane & 15;

    v8f acc[4];
    #pragma unroll
    for (int i = 0; i < 4; ++i)
        #pragma unroll
        for (int e = 0; e < 8; ++e) acc[i][e] = 0.0f;

    for (int kt = 0; kt < CDIM; kt += 32) {
        for (int idx = tid; idx < 64 * 32; idx += 128) {       // A: coalesced along C
            int k = idx & 31, m = idx >> 5;
            As[m * PITCH + k] = f2bf_bits(Wp[(size_t)(o0 + m) * CDIM + kt + k]);
        }
        for (int idx = tid; idx < 64 * 32; idx += 128) {       // B: coalesced along N
            int n = idx & 63, k = idx >> 6;
            Bs[n * PITCH + k] = f2bf_bits(src[(size_t)(kt + k) * NDIM + n0 + n]);
        }
        __syncthreads();

        v16bf afrag = load_frag(As, (wv * 16 + mr) * PITCH, g);
        #pragma unroll
        for (int nn = 0; nn < 4; ++nn) {
            v16bf bfrag = load_frag(Bs, (nn * 16 + mr) * PITCH, g);
            acc[nn] = __builtin_amdgcn_wmma_f32_16x16x32_bf16(
                false, afrag, false, bfrag, (short)0, acc[nn], false, false);
        }
        __syncthreads();
    }

    if (p == 2) {          // hx: row-major (C x N) direct stores
        #pragma unroll
        for (int nn = 0; nn < 4; ++nn)
            #pragma unroll
            for (int r = 0; r < 8; ++r) {
                int row = o0 + wv * 16 + r + 8 * g;
                int col = n0 + nn * 16 + mr;
                dst[(size_t)row * NDIM + col] = f2bf_bits(acc[nn][r] + bias[row]);
            }
    } else {               // fwT/gxT: transpose through LDS, coalesced (N x C2) stores
        unsigned short* stage = smem;     // 64x64 tile, pitch 66 (bank stagger)
        #pragma unroll
        for (int nn = 0; nn < 4; ++nn)
            #pragma unroll
            for (int r = 0; r < 8; ++r) {
                int row = wv * 16 + r + 8 * g;          // local o
                int col = nn * 16 + mr;                 // local n
                stage[col * 66 + row] = f2bf_bits(acc[nn][r] + bias[o0 + row]);
            }
        __syncthreads();
        for (int idx = tid; idx < 64 * 64; idx += 128) {
            int o = idx & 63, n = idx >> 6;
            dst[(size_t)(n0 + n) * C2DIM + o0 + o] = stage[n * 66 + o];
        }
    }
}

// ---------------------------------------------------------------------------
// Kernel 2: transposed logits  L[b,j,i] = sum_c g_x[c,j] * f_w[c,i]
// Both operands are (N x C2) bf16 -> double-buffered async global->LDS copies
// overlap the HBM fetch of tile t+1 with the WMMAs on tile t.
// ---------------------------------------------------------------------------
__global__ __launch_bounds__(128)
void logits_kernel(const unsigned short* __restrict__ fwT,
                   const unsigned short* __restrict__ gxT,
                   float* __restrict__ attn)
{
    const int i0 = blockIdx.x * 64;
    const int j0 = blockIdx.y * 64;
    const int b  = blockIdx.z;
    const unsigned short* fb = fwT + (size_t)b * C2DIM * NDIM;
    const unsigned short* gb = gxT + (size_t)b * C2DIM * NDIM;

    __shared__ __align__(16) unsigned short smem[4 * TILEH];   // {A,B} x 2 buffers
    const unsigned sbase = lds_off(smem);

    const int tid  = threadIdx.x;
    const int lane = tid & 31;
    const int wv   = tid >> 5;
    const int g    = lane >> 4;
    const int mr   = lane & 15;

    v8f acc[4];
    #pragma unroll
    for (int i = 0; i < 4; ++i)
        #pragma unroll
        for (int e = 0; e < 8; ++e) acc[i][e] = 0.0f;

    // issue async copies of both 64x32 bf16 tiles for k-step kt into buffer buf
    auto issue = [&](int kt, int buf) {
        #pragma unroll
        for (int s = 0; s < 2; ++s) {
            int q  = tid + s * 128;          // 256 16B chunks per tile
            int m  = q >> 2;
            int hf = (q & 3) * 8;
            async_b128(sbase + (unsigned)(buf * 2 * TILEH + m * PITCH + hf) * 2,
                       gb + (size_t)(j0 + m) * C2DIM + kt + hf);
            async_b128(sbase + (unsigned)(buf * 2 * TILEH + TILEH + m * PITCH + hf) * 2,
                       fb + (size_t)(i0 + m) * C2DIM + kt + hf);
        }
    };

    issue(0, 0);
    for (int t = 0; t < C2DIM / 32; ++t) {
        const int cur = t & 1;
        if (t + 1 < C2DIM / 32) {
            issue((t + 1) * 32, cur ^ 1);
            asm volatile("s_wait_asynccnt 0x4" ::: "memory");  // 4 newest in flight
        } else {
            asm volatile("s_wait_asynccnt 0x0" ::: "memory");
        }
        __syncthreads();

        const unsigned short* Ac = smem + cur * 2 * TILEH;
        const unsigned short* Bc = Ac + TILEH;
        v16bf afrag = load_frag(Ac, (wv * 16 + mr) * PITCH, g);
        #pragma unroll
        for (int nn = 0; nn < 4; ++nn) {
            v16bf bfrag = load_frag(Bc, (nn * 16 + mr) * PITCH, g);
            acc[nn] = __builtin_amdgcn_wmma_f32_16x16x32_bf16(
                false, afrag, false, bfrag, (short)0, acc[nn], false, false);
        }
        __syncthreads();
    }

    #pragma unroll
    for (int nn = 0; nn < 4; ++nn)
        #pragma unroll
        for (int r = 0; r < 8; ++r) {
            int row = j0 + wv * 16 + r + 8 * g;
            int col = i0 + nn * 16 + mr;
            attn[(size_t)b * NDIM * NDIM + (size_t)row * NDIM + col] = acc[nn][r];
        }
}

// ---------------------------------------------------------------------------
// Kernel 3: in-place softmax along contiguous rows of attn_T (length 2304).
// ---------------------------------------------------------------------------
__global__ __launch_bounds__(256)
void softmax_kernel(float* __restrict__ attn)
{
    float* p = attn + (size_t)blockIdx.x * NDIM;
    __shared__ float red[256];
    const int tid = threadIdx.x;

    float m = -1e30f;
    for (int i = tid; i < NDIM; i += 256) m = fmaxf(m, p[i]);
    red[tid] = m; __syncthreads();
    for (int s = 128; s > 0; s >>= 1) {
        if (tid < s) red[tid] = fmaxf(red[tid], red[tid + s]);
        __syncthreads();
    }
    m = red[0]; __syncthreads();

    float sum = 0.0f;
    for (int i = tid; i < NDIM; i += 256) {
        float e = __expf(p[i] - m);
        p[i] = e;
        sum += e;
    }
    red[tid] = sum; __syncthreads();
    for (int s = 128; s > 0; s >>= 1) {
        if (tid < s) red[tid] += red[tid + s];
        __syncthreads();
    }
    float inv = 1.0f / red[0];
    for (int i = tid; i < NDIM; i += 256) p[i] *= inv;
}

// ---------------------------------------------------------------------------
// Kernel 4: out[b,c,i] = gamma * sum_j h_x[c,j] * attnT[j,i] + w[b,c,i]
// h_x tiles double-buffered via async copies; attention tiles need fp32->bf16
// conversion so they go through the VALU path (+ prefetch hint).
// ---------------------------------------------------------------------------
__global__ __launch_bounds__(128)
void av_kernel(const unsigned short* __restrict__ hx,
               const float* __restrict__ attn,
               const float* __restrict__ w,
               const float* __restrict__ gammaPtr,
               float* __restrict__ out)
{
    const int i0 = blockIdx.x * 64;
    const int c0 = blockIdx.y * 64;
    const int b  = blockIdx.z;
    const unsigned short* hxb  = hx   + (size_t)b * CDIM * NDIM;
    const float*          attb = attn + (size_t)b * NDIM * NDIM;

    __shared__ __align__(16) unsigned short smem[3 * TILEH];   // A x 2 buffers | B
    const unsigned sbase = lds_off(smem);
    unsigned short* Bb = smem + 2 * TILEH;

    const int tid  = threadIdx.x;
    const int lane = tid & 31;
    const int wv   = tid >> 5;
    const int g    = lane >> 4;
    const int mr   = lane & 15;

    v8f acc[4];
    #pragma unroll
    for (int i = 0; i < 4; ++i)
        #pragma unroll
        for (int e = 0; e < 8; ++e) acc[i][e] = 0.0f;

    auto issueA = [&](int kt, int buf) {
        #pragma unroll
        for (int s = 0; s < 2; ++s) {
            int q  = tid + s * 128;
            int m  = q >> 2;
            int hf = (q & 3) * 8;
            async_b128(sbase + (unsigned)(buf * TILEH + m * PITCH + hf) * 2,
                       hxb + (size_t)(c0 + m) * NDIM + kt + hf);
        }
    };

    issueA(0, 0);
    for (int t = 0; t < NDIM / 32; ++t) {
        const int cur = t & 1;
        const int kt  = t * 32;
        if (t + 1 < NDIM / 32) issueA(kt + 32, cur ^ 1);

        for (int idx = tid; idx < 64 * 32; idx += 128) {   // attn tile: fp32 -> bf16
            int n = idx & 63, k = idx >> 6;
            Bb[n * PITCH + k] = f2bf_bits(attb[(size_t)(kt + k) * NDIM + i0 + n]);
        }
        if (t + 1 < NDIM / 32) {
            __builtin_prefetch(&attb[(size_t)(kt + 32) * NDIM + i0 + (tid & 63)], 0, 1);
            asm volatile("s_wait_asynccnt 0x2" ::: "memory");
        } else {
            asm volatile("s_wait_asynccnt 0x0" ::: "memory");
        }
        __syncthreads();

        const unsigned short* Ac = smem + cur * TILEH;
        v16bf afrag = load_frag(Ac, (wv * 16 + mr) * PITCH, g);
        #pragma unroll
        for (int nn = 0; nn < 4; ++nn) {
            v16bf bfrag = load_frag(Bb, (nn * 16 + mr) * PITCH, g);
            acc[nn] = __builtin_amdgcn_wmma_f32_16x16x32_bf16(
                false, afrag, false, bfrag, (short)0, acc[nn], false, false);
        }
        __syncthreads();
    }

    const float gamma = gammaPtr[0];
    #pragma unroll
    for (int nn = 0; nn < 4; ++nn)
        #pragma unroll
        for (int r = 0; r < 8; ++r) {
            int row = c0 + wv * 16 + r + 8 * g;
            int col = i0 + nn * 16 + mr;
            size_t o = (size_t)b * CDIM * NDIM + (size_t)row * NDIM + col;
            out[o] = gamma * acc[nn][r] + w[o];
        }
}

// ---------------------------------------------------------------------------
extern "C" void kernel_launch(void* const* d_in, const int* in_sizes, int n_in,
                              void* d_out, int out_size, void* d_ws, size_t ws_size,
                              hipStream_t stream)
{
    const float* x     = (const float*)d_in[0];
    const float* w     = (const float*)d_in[1];
    const float* Wf    = (const float*)d_in[2];
    const float* bf    = (const float*)d_in[3];
    const float* Wg    = (const float*)d_in[4];
    const float* bg    = (const float*)d_in[5];
    const float* Wh    = (const float*)d_in[6];
    const float* bh    = (const float*)d_in[7];
    const float* gamma = (const float*)d_in[8];

    float* out  = (float*)d_out;
    float* attn = out + (size_t)BDIM * CDIM * NDIM;   // attn_T region of d_out

    unsigned short* fwT = (unsigned short*)d_ws;                   // (B,N,C2) bf16
    unsigned short* gxT = fwT + (size_t)BDIM * C2DIM * NDIM;       // (B,N,C2) bf16
    unsigned short* hx  = gxT + (size_t)BDIM * C2DIM * NDIM;       // (B,C,N)  bf16

    dim3 blk(128, 1, 1);
    proj_kernel  <<<dim3(NDIM / 64, CDIM / 64, BDIM * 3), blk, 0, stream>>>(
        x, w, Wf, bf, Wg, bg, Wh, bh, fwT, gxT, hx);
    logits_kernel<<<dim3(NDIM / 64, NDIM / 64, BDIM), blk, 0, stream>>>(fwT, gxT, attn);
    softmax_kernel<<<dim3(BDIM * NDIM, 1, 1), dim3(256, 1, 1), 0, stream>>>(attn);
    av_kernel    <<<dim3(NDIM / 64, CDIM / 64, BDIM), blk, 0, stream>>>(
        hx, attn, w, gamma, out);
}
